// RouteNet_Fermi_8504035246172
// MI455X (gfx1250) — compile-verified, matrix operand order checked by reading, not run on hardware
//
#include <hip/hip_runtime.h>
#include <hip/hip_bf16.h>
#include <stdint.h>

// ---------------------------------------------------------------------------
// RouteNet-Fermi on MI455X (gfx1250): wave32 WMMA f16 tiles, f16 state mirrors
// Round 4: async global<->LDS fixed to the builtin's real signature
//          (addrspace(1) int4*, addrspace(3) int4*, imm offset, cpol)
// ---------------------------------------------------------------------------

typedef __attribute__((ext_vector_type(16))) _Float16 v16h;
typedef __attribute__((ext_vector_type(8)))  _Float16 v8h;
typedef __attribute__((ext_vector_type(8)))  float    v8f;
typedef __attribute__((ext_vector_type(4)))  int      v4i;

constexpr int P    = 50000;
constexpr int NL   = 2000;
constexpr int NQ   = 6000;
constexpr int L    = 8;
constexpr int P2L  = 128;
constexpr int P2Q  = 64;
constexpr int D    = 32;
constexpr int MAX_MODELS = 7;
constexpr int NPOL = 4;
constexpr int MAXQ = 3;
constexpr int ITERS = 8;
constexpr int WAVES = 4;          // waves (16-row tiles) per 128-thread block

constexpr int PTILES = P  / 16;   // 3125
constexpr int LTILES = NL / 16;   // 125
constexpr int QTILES = NQ / 16;   // 375
constexpr int PSS_STRIDE = (L + 1) * D;   // 288 halves per path

// --------------------------- WMMA fragment helpers -------------------------
__device__ __forceinline__ int wmma_koff(int v, int half) {
  return (v < 4 ? 2 * v : 16 + 2 * (v - 4)) + 8 * half;
}

// per lane the 16 A elements are two contiguous 8-half runs: [8h,8h+8) and
// [16+8h, 16+8h+8). Load each run as one 16-byte LDS access (ds_load_b128).
__device__ __forceinline__ v16h load_a_frag(const _Float16* A, int lane) {
  int m = lane & 15, hh = lane >> 4;
  v8h lo = *(const v8h*)(A + m * 32 + 8 * hh);
  v8h hi = *(const v8h*)(A + m * 32 + 16 + 8 * hh);
  return __builtin_shufflevector(lo, hi, 0, 1, 2, 3, 4, 5, 6, 7,
                                 8, 9, 10, 11, 12, 13, 14, 15);
}

// B fragments staged pre-swizzled: Bswz[blk*512 + lane*16 + j]; load = 2x b128.
__device__ __forceinline__ v16h load_bswz(const _Float16* Bswz, int blk, int lane) {
  const _Float16* p = Bswz + blk * 512 + lane * 16;
  v8h lo = *(const v8h*)p;
  v8h hi = *(const v8h*)(p + 8);
  return __builtin_shufflevector(lo, hi, 0, 1, 2, 3, 4, 5, 6, 7,
                                 8, 9, 10, 11, 12, 13, 14, 15);
}

__device__ __forceinline__ v8f wmma16(v16h a, v16h b, v8f c) {
  return __builtin_amdgcn_wmma_f32_16x16x32_f16(
      /*neg_a=*/false, a, /*neg_b=*/false, b,
      /*c_mod=*/(short)0, c, /*reuse_a=*/false, /*reuse_b=*/false);
}

__device__ __forceinline__ v8f zero8() {
  v8f z = {0.f, 0.f, 0.f, 0.f, 0.f, 0.f, 0.f, 0.f};
  return z;
}

// stage f32 weight (kin x srcld) into swizzled-fragment LDS layout.
__device__ __forceinline__ void stage_bfrag(const float* __restrict__ W,
                                            _Float16* dst, int kin, int srcld,
                                            int nblk, int tid, int nth) {
  for (int i = tid; i < nblk * 512; i += nth) {
    int blk  = i >> 9;
    int lane = (i >> 4) & 31;
    int j    = i & 15;
    int hh   = lane >> 4;
    int k    = wmma_koff(j >> 1, hh) + (j & 1);
    int n    = (lane & 15) + 16 * blk;
    dst[i] = (k < kin) ? (_Float16)W[k * srcld + n] : (_Float16)0.f;
  }
}

// copy one 16-half (32-byte) row slice through VGPRs: two 128-bit transfers
__device__ __forceinline__ void cprow(_Float16* dst, const _Float16* src) {
  v8h a0 = *(const v8h*)src;
  v8h a1 = *(const v8h*)(src + 8);
  *(v8h*)dst = a0;
  *(v8h*)(dst + 8) = a1;
}

// ------------------- CDNA5 async global<->LDS row movers --------------------
#if defined(__has_builtin)
#if __has_builtin(__builtin_amdgcn_global_load_async_to_lds_b128) && \
    __has_builtin(__builtin_amdgcn_global_store_async_from_lds_b128) && \
    __has_builtin(__builtin_amdgcn_s_wait_asynccnt)
#define ASYNC_LDS 1
#endif
#endif

#ifdef ASYNC_LDS
typedef __attribute__((address_space(1))) v4i* gp128;   // global int4*
typedef __attribute__((address_space(3))) v4i* lp128;   // LDS int4*
// global -> LDS, 32 bytes per lane, tracked by ASYNCcnt
__device__ __forceinline__ void g2l_row(_Float16* lds_dst, const _Float16* gsrc) {
  __builtin_amdgcn_global_load_async_to_lds_b128(
      (gp128)gsrc, (lp128)lds_dst, 0, 0);
  __builtin_amdgcn_global_load_async_to_lds_b128(
      (gp128)(gsrc + 8), (lp128)(lds_dst + 8), 0, 0);
}
// LDS -> global, 32 bytes per lane, tracked by ASYNCcnt
__device__ __forceinline__ void l2g_row(_Float16* gdst, const _Float16* lds_src) {
  __builtin_amdgcn_global_store_async_from_lds_b128(
      (gp128)gdst, (lp128)lds_src, 0, 0);
  __builtin_amdgcn_global_store_async_from_lds_b128(
      (gp128)(gdst + 8), (lp128)(lds_src + 8), 0, 0);
}
__device__ __forceinline__ void async_fence() {
  __builtin_amdgcn_s_wait_asynccnt(0);
}
#else
__device__ __forceinline__ void g2l_row(_Float16* lds_dst, const _Float16* gsrc) {
  cprow(lds_dst, gsrc);
}
__device__ __forceinline__ void l2g_row(_Float16* gdst, const _Float16* lds_src) {
  cprow(gdst, lds_src);
}
__device__ __forceinline__ void async_fence() {}
#endif

// ---- fused two-layer relu MLP on one 16x32 tile (result left in H, f16) ----
__device__ __forceinline__ void mlp2_tile(const _Float16* X, const _Float16* W1f,
                                          const float* B1s, const _Float16* W2f,
                                          const float* B2s, _Float16* H, int lane) {
  int n = lane & 15, hh = lane >> 4;
  v16h a = load_a_frag(X, lane);
  v8f c0 = wmma16(a, load_bswz(W1f, 0, lane), zero8());
  v8f c1 = wmma16(a, load_bswz(W1f, 1, lane), zero8());
#pragma unroll
  for (int v = 0; v < 8; ++v) {
    int m = v + 8 * hh;
    float x0 = c0[v] + B1s[n];
    float x1 = c1[v] + B1s[16 + n];
    H[m * 32 + n]      = (_Float16)(x0 > 0.f ? x0 : 0.f);
    H[m * 32 + 16 + n] = (_Float16)(x1 > 0.f ? x1 : 0.f);
  }
  __syncthreads();
  a  = load_a_frag(H, lane);
  c0 = wmma16(a, load_bswz(W2f, 0, lane), zero8());
  c1 = wmma16(a, load_bswz(W2f, 1, lane), zero8());
  __syncthreads();
#pragma unroll
  for (int v = 0; v < 8; ++v) {
    int m = v + 8 * hh;
    float x0 = c0[v] + B2s[n];
    float x1 = c1[v] + B2s[16 + n];
    H[m * 32 + n]      = (_Float16)(x0 > 0.f ? x0 : 0.f);
    H[m * 32 + 16 + n] = (_Float16)(x1 > 0.f ? x1 : 0.f);
  }
  __syncthreads();
}

// --------------------------- kernel: link load -----------------------------
__global__ __launch_bounds__(128) void k_load(const float* __restrict__ traffic,
                                              const int* __restrict__ p2l,
                                              const int* __restrict__ p2l_deg,
                                              const float* __restrict__ capacity,
                                              float* __restrict__ loadv) {
  __shared__ float s[128];
  int l = blockIdx.x, t = threadIdx.x;
  float v = 0.f;
  if (t < p2l_deg[l]) v = traffic[p2l[(l * P2L + t) * 2]];
  s[t] = v;
  __syncthreads();
  for (int st = 64; st > 0; st >>= 1) {
    if (t < st) s[t] += s[t + st];
    __syncthreads();
  }
  if (t == 0) loadv[l] = s[0] / capacity[l];
}

// --------------------------- kernel: path embedding ------------------------
__global__ __launch_bounds__(128) void k_path_embed(
    const float* __restrict__ traffic, const float* __restrict__ packets,
    const float* __restrict__ eq_lambda, const float* __restrict__ apl,
    const float* __restrict__ emf, const float* __restrict__ plo,
    const float* __restrict__ atoff, const float* __restrict__ aton,
    const float* __restrict__ ar_a, const float* __restrict__ sigma,
    const int* __restrict__ model, const float* __restrict__ W1,
    const float* __restrict__ B1, const float* __restrict__ W2,
    const float* __restrict__ B2, _Float16* __restrict__ ps16) {
  __shared__ __align__(16) _Float16 W1f[1024], W2f[1024];
  __shared__ float B1s[32], B2s[32];
  __shared__ __align__(16) _Float16 Xs[WAVES][512], Hs[WAVES][512];
  int tid = threadIdx.x, lane = tid & 31, wv = tid >> 5;
  stage_bfrag(W1, W1f, 10 + MAX_MODELS, 32, 2, tid, blockDim.x);
  stage_bfrag(W2, W2f, 32, 32, 2, tid, blockDim.x);
  if (tid < 32) { B1s[tid] = B1[tid]; B2s[tid] = B2[tid]; }
  int tile = blockIdx.x * WAVES + wv;
  bool active = tile < PTILES;
  if (!active) tile = 0;
  int p0 = tile * 16;
  _Float16* X = Xs[wv];
  _Float16* H = Hs[wv];
  if (lane < 16) {
    int p = p0 + lane;
    float f[17];
    f[0] = (traffic[p] - 1385.4059f) / 859.8119f;
    f[1] = (packets[p] - 1.4015f) / 0.8933f;
#pragma unroll
    for (int i = 0; i < MAX_MODELS; ++i) f[2 + i] = 0.f;
    f[2 + model[p]] = 1.f;
    f[9]  = (eq_lambda[p] - 1350.9712f) / 858.3162f;
    f[10] = (apl[p] - 0.9117f) / 0.9724f;
    f[11] = (emf[p] - 6.6636f) / 4.7151f;
    f[12] = (plo[p] - 0.9116f) / 1.6513f;
    f[13] = (atoff[p] - 1.6649f) / 2.3564f;
    f[14] = (aton[p] - 1.6649f) / 2.3564f;
    f[15] = ar_a[p];
    f[16] = sigma[p];
#pragma unroll
    for (int j = 0; j < 32; ++j)
      X[lane * 32 + j] = (_Float16)(j < 17 ? f[j] : 0.f);
  }
  __syncthreads();
  mlp2_tile(X, W1f, B1s, W2f, B2s, H, lane);
  int r = lane & 15, hh = lane >> 4;
  if (active)
    l2g_row(ps16 + (p0 + r) * D + hh * 16, H + r * 32 + hh * 16);
}

// --------------------------- kernel: link embedding ------------------------
__global__ __launch_bounds__(128) void k_link_embed(
    const float* __restrict__ loadv, const int* __restrict__ policy,
    const float* __restrict__ W1, const float* __restrict__ B1,
    const float* __restrict__ W2, const float* __restrict__ B2,
    _Float16* __restrict__ ls16) {
  __shared__ __align__(16) _Float16 W1f[1024], W2f[1024];
  __shared__ float B1s[32], B2s[32];
  __shared__ __align__(16) _Float16 Xs[WAVES][512], Hs[WAVES][512];
  int tid = threadIdx.x, lane = tid & 31, wv = tid >> 5;
  stage_bfrag(W1, W1f, NPOL + 1, 32, 2, tid, blockDim.x);
  stage_bfrag(W2, W2f, 32, 32, 2, tid, blockDim.x);
  if (tid < 32) { B1s[tid] = B1[tid]; B2s[tid] = B2[tid]; }
  int tile = blockIdx.x * WAVES + wv;
  bool active = tile < LTILES;
  if (!active) tile = 0;
  int l0 = tile * 16;
  _Float16* X = Xs[wv];
  _Float16* H = Hs[wv];
  if (lane < 16) {
    int l = l0 + lane;
    float f[5] = {loadv[l], 0.f, 0.f, 0.f, 0.f};
    f[1 + policy[l]] = 1.f;
#pragma unroll
    for (int j = 0; j < 32; ++j)
      X[lane * 32 + j] = (_Float16)(j < 5 ? f[j] : 0.f);
  }
  __syncthreads();
  mlp2_tile(X, W1f, B1s, W2f, B2s, H, lane);
  int r = lane & 15, hh = lane >> 4;
  if (active)
    l2g_row(ls16 + (l0 + r) * D + hh * 16, H + r * 32 + hh * 16);
}

// --------------------------- kernel: queue embedding -----------------------
__global__ __launch_bounds__(128) void k_queue_embed(
    const float* __restrict__ qsize, const int* __restrict__ priority,
    const float* __restrict__ weight, const float* __restrict__ W1,
    const float* __restrict__ B1, const float* __restrict__ W2,
    const float* __restrict__ B2, _Float16* __restrict__ qs16) {
  __shared__ __align__(16) _Float16 W1f[1024], W2f[1024];
  __shared__ float B1s[32], B2s[32];
  __shared__ __align__(16) _Float16 Xs[WAVES][512], Hs[WAVES][512];
  int tid = threadIdx.x, lane = tid & 31, wv = tid >> 5;
  stage_bfrag(W1, W1f, MAXQ + 2, 32, 2, tid, blockDim.x);
  stage_bfrag(W2, W2f, 32, 32, 2, tid, blockDim.x);
  if (tid < 32) { B1s[tid] = B1[tid]; B2s[tid] = B2[tid]; }
  int tile = blockIdx.x * WAVES + wv;
  bool active = tile < QTILES;
  if (!active) tile = 0;
  int q0 = tile * 16;
  _Float16* X = Xs[wv];
  _Float16* H = Hs[wv];
  if (lane < 16) {
    int q = q0 + lane;
    float f[5] = {(qsize[q] - 30259.1055f) / 21410.0957f, 0.f, 0.f, 0.f, 0.f};
    f[1 + priority[q]] = 1.f;
    f[4] = weight[q];
#pragma unroll
    for (int j = 0; j < 32; ++j)
      X[lane * 32 + j] = (_Float16)(j < 5 ? f[j] : 0.f);
  }
  __syncthreads();
  mlp2_tile(X, W1f, B1s, W2f, B2s, H, lane);
  int r = lane & 15, hh = lane >> 4;
  if (active)
    l2g_row(qs16 + (q0 + r) * D + hh * 16, H + r * 32 + hh * 16);
}

// --------------------------- kernel: path RNN (per iteration) --------------
__global__ __launch_bounds__(128) void k_path_rnn(
    _Float16* __restrict__ ps16, const _Float16* __restrict__ qs16,
    const _Float16* __restrict__ ls16, const int* __restrict__ qtp,
    const int* __restrict__ ltp, const int* __restrict__ length,
    const float* __restrict__ prW, const float* __restrict__ prU,
    const float* __restrict__ prB, _Float16* __restrict__ pss16) {
  __shared__ __align__(16) _Float16 Wqf[1024], Wlf[1024], Uf[1024];
  __shared__ float Bs[32];
  __shared__ __align__(16) _Float16 Hh[WAVES][512], Xq[WAVES][512], Xl[WAVES][512];
  __shared__ int lenS[WAVES][16];
  int tid = threadIdx.x, lane = tid & 31, wv = tid >> 5;
  stage_bfrag(prW,        Wqf, 32, 32, 2, tid, blockDim.x);  // rows 0..31 (queue)
  stage_bfrag(prW + 1024, Wlf, 32, 32, 2, tid, blockDim.x);  // rows 32..63 (link)
  stage_bfrag(prU,        Uf,  32, 32, 2, tid, blockDim.x);
  if (tid < 32) Bs[tid] = prB[tid];
  int tile = blockIdx.x * WAVES + wv;
  bool active = tile < PTILES;
  if (!active) tile = 0;
  int p0 = tile * 16;
  _Float16* h  = Hh[wv];
  _Float16* xq = Xq[wv];
  _Float16* xl = Xl[wv];
  int r = lane & 15, hh = lane >> 4, n = r;
  if (lane < 16) lenS[wv][lane] = length[p0 + lane];
  // load h tile (f16) and emit pss[:,0] = prev state (128-bit transfers)
  {
    const _Float16* src = ps16 + (p0 + r) * D + hh * 16;
    v8h a0 = *(const v8h*)src;
    v8h a1 = *(const v8h*)(src + 8);
    *(v8h*)(h + r * 32 + hh * 16) = a0;
    *(v8h*)(h + r * 32 + hh * 16 + 8) = a1;
    if (active) {
      _Float16* d = pss16 + (size_t)(p0 + r) * PSS_STRIDE + hh * 16;
      *(v8h*)d = a0;
      *(v8h*)(d + 8) = a1;
    }
  }
  __syncthreads();
  for (int t = 0; t < L; ++t) {
    int qi = qtp[(p0 + r) * L + t];
    int li = ltp[(p0 + r) * L + t];
    g2l_row(xq + r * 32 + hh * 16, qs16 + qi * D + hh * 16);
    g2l_row(xl + r * 32 + hh * 16, ls16 + li * D + hh * 16);
    if (t + 1 < L) {   // prefetch next hop's gather rows (global_prefetch_b8)
      int qn = qtp[(p0 + r) * L + t + 1];
      int ln = ltp[(p0 + r) * L + t + 1];
      __builtin_prefetch(qs16 + qn * D + hh * 16, 0, 1);
      __builtin_prefetch(ls16 + ln * D + hh * 16, 0, 1);
    }
    async_fence();     // xq/xl landed; also prev step's pss store done (h reuse)
    __syncthreads();
    v8f c0 = wmma16(load_a_frag(xq, lane), load_bswz(Wqf, 0, lane), zero8());
    c0 = wmma16(load_a_frag(xl, lane), load_bswz(Wlf, 0, lane), c0);
    c0 = wmma16(load_a_frag(h,  lane), load_bswz(Uf,  0, lane), c0);
    v8f c1 = wmma16(load_a_frag(xq, lane), load_bswz(Wqf, 1, lane), zero8());
    c1 = wmma16(load_a_frag(xl, lane), load_bswz(Wlf, 1, lane), c1);
    c1 = wmma16(load_a_frag(h,  lane), load_bswz(Uf,  1, lane), c1);
    __syncthreads();  // all reads of h done before overwrite
#pragma unroll
    for (int v = 0; v < 8; ++v) {
      int m = v + 8 * hh;
      bool on = t < lenS[wv][m];
      float h0 = on ? tanhf(c0[v] + Bs[n])      : (float)h[m * 32 + n];
      float h1 = on ? tanhf(c1[v] + Bs[16 + n]) : (float)h[m * 32 + 16 + n];
      h[m * 32 + n]      = (_Float16)h0;
      h[m * 32 + 16 + n] = (_Float16)h1;
    }
    __syncthreads();
    if (active)
      l2g_row(pss16 + (size_t)(p0 + r) * PSS_STRIDE + (t + 1) * D + hh * 16,
              h + r * 32 + hh * 16);
  }
  if (active)
    l2g_row(ps16 + (p0 + r) * D + hh * 16, h + r * 32 + hh * 16);
  // S_ENDPGM performs an implicit wait-idle, draining outstanding async stores.
}

// --------------------------- kernel: queue update --------------------------
__global__ __launch_bounds__(128) void k_queue_update(
    const _Float16* __restrict__ pss16, _Float16* __restrict__ qs16,
    const int* __restrict__ p2q, const int* __restrict__ p2q_deg,
    const float* __restrict__ qrW, const float* __restrict__ qrU,
    const float* __restrict__ qrB) {
  __shared__ __align__(16) _Float16 Wf[1024], Uf[1024];
  __shared__ float Bs[32];
  __shared__ __align__(16) _Float16 As[WAVES][512], Qs[WAVES][512];
  int tid = threadIdx.x, lane = tid & 31, wv = tid >> 5;
  stage_bfrag(qrW, Wf, 32, 32, 2, tid, blockDim.x);
  stage_bfrag(qrU, Uf, 32, 32, 2, tid, blockDim.x);
  if (tid < 32) Bs[tid] = qrB[tid];
  int tile = blockIdx.x * WAVES + wv;
  bool active = tile < QTILES;
  if (!active) tile = 0;
  int q0 = tile * 16;
  int r = lane & 15, hh = lane >> 4, n = r;
  // queue-state row arrives via async DMA while we aggregate path states
  g2l_row(Qs[wv] + r * 32 + hh * 16, qs16 + (q0 + r) * D + hh * 16);
  int deg = p2q_deg[q0 + r];
  float acc[16];
#pragma unroll
  for (int j = 0; j < 16; ++j) acc[j] = 0.f;
  for (int e = 0; e < P2Q; ++e) {
    if (e < deg) {
      int pe  = p2q[((q0 + r) * P2Q + e) * 2];
      int pos = p2q[((q0 + r) * P2Q + e) * 2 + 1];
      const v8h* src = (const v8h*)(pss16 + (size_t)pe * PSS_STRIDE + pos * D + hh * 16);
      v8h x0 = src[0];
      v8h x1 = src[1];
      if (e + 2 < deg) {  // lookahead prefetch
        int pn  = p2q[((q0 + r) * P2Q + e + 2) * 2];
        int psn = p2q[((q0 + r) * P2Q + e + 2) * 2 + 1];
        __builtin_prefetch(pss16 + (size_t)pn * PSS_STRIDE + psn * D + hh * 16, 0, 1);
      }
#pragma unroll
      for (int j = 0; j < 8; ++j) {
        acc[j]     += (float)x0[j];
        acc[8 + j] += (float)x1[j];
      }
    }
  }
  {
    v8h y0, y1;
#pragma unroll
    for (int j = 0; j < 8; ++j) {
      y0[j] = (_Float16)acc[j];
      y1[j] = (_Float16)acc[8 + j];
    }
    *(v8h*)(As[wv] + r * 32 + hh * 16) = y0;
    *(v8h*)(As[wv] + r * 32 + hh * 16 + 8) = y1;
  }
  async_fence();
  __syncthreads();
  v8f c0 = wmma16(load_a_frag(As[wv], lane), load_bswz(Wf, 0, lane), zero8());
  c0 = wmma16(load_a_frag(Qs[wv], lane), load_bswz(Uf, 0, lane), c0);
  v8f c1 = wmma16(load_a_frag(As[wv], lane), load_bswz(Wf, 1, lane), zero8());
  c1 = wmma16(load_a_frag(Qs[wv], lane), load_bswz(Uf, 1, lane), c1);
  __syncthreads();
#pragma unroll
  for (int v = 0; v < 8; ++v) {
    int m = v + 8 * hh;
    As[wv][m * 32 + n]      = (_Float16)tanhf(c0[v] + Bs[n]);
    As[wv][m * 32 + 16 + n] = (_Float16)tanhf(c1[v] + Bs[16 + n]);
  }
  __syncthreads();
  if (active)
    l2g_row(qs16 + (q0 + r) * D + hh * 16, As[wv] + r * 32 + hh * 16);
}

// --------------------------- kernel: link update ---------------------------
__global__ __launch_bounds__(128) void k_link_update(
    const _Float16* __restrict__ qs16, _Float16* __restrict__ ls16,
    const int* __restrict__ qtl, const float* __restrict__ lrW,
    const float* __restrict__ lrU, const float* __restrict__ lrB) {
  __shared__ __align__(16) _Float16 Wf[1024], Uf[1024];
  __shared__ float Bs[32];
  __shared__ __align__(16) _Float16 Hh[WAVES][512], Xs[WAVES][512];
  int tid = threadIdx.x, lane = tid & 31, wv = tid >> 5;
  stage_bfrag(lrW, Wf, 32, 32, 2, tid, blockDim.x);
  stage_bfrag(lrU, Uf, 32, 32, 2, tid, blockDim.x);
  if (tid < 32) Bs[tid] = lrB[tid];
  int tile = blockIdx.x * WAVES + wv;
  bool active = tile < LTILES;
  if (!active) tile = 0;
  int l0 = tile * 16;
  _Float16* h = Hh[wv];
  _Float16* x = Xs[wv];
  int r = lane & 15, hh = lane >> 4, n = r;
  cprow(h + r * 32 + hh * 16, ls16 + (l0 + r) * D + hh * 16);
  __syncthreads();
  for (int t = 0; t < MAXQ; ++t) {
    int qi = qtl[(l0 + r) * MAXQ + t];
    g2l_row(x + r * 32 + hh * 16, qs16 + qi * D + hh * 16);
    async_fence();
    __syncthreads();
    v8f c0 = wmma16(load_a_frag(x, lane), load_bswz(Wf, 0, lane), zero8());
    c0 = wmma16(load_a_frag(h, lane), load_bswz(Uf, 0, lane), c0);
    v8f c1 = wmma16(load_a_frag(x, lane), load_bswz(Wf, 1, lane), zero8());
    c1 = wmma16(load_a_frag(h, lane), load_bswz(Uf, 1, lane), c1);
    __syncthreads();
#pragma unroll
    for (int v = 0; v < 8; ++v) {
      int m = v + 8 * hh;
      h[m * 32 + n]      = (_Float16)tanhf(c0[v] + Bs[n]);
      h[m * 32 + 16 + n] = (_Float16)tanhf(c1[v] + Bs[16 + n]);
    }
    __syncthreads();
  }
  if (active)
    l2g_row(ls16 + (l0 + r) * D + hh * 16, h + r * 32 + hh * 16);
}

// --------------------------- kernel: readout -------------------------------
__global__ __launch_bounds__(128) void k_readout(
    const _Float16* __restrict__ pss16, const int* __restrict__ ltp,
    const int* __restrict__ length, const float* __restrict__ capacity,
    const float* __restrict__ traffic, const float* __restrict__ packets,
    const float* __restrict__ roW1, const float* __restrict__ roB1,
    const float* __restrict__ roW2, const float* __restrict__ roB2,
    const float* __restrict__ roW3, const float* __restrict__ roB3,
    float* __restrict__ out) {
  __shared__ __align__(16) _Float16 W1f[512];   // swizzled, 1 column block
  __shared__ __align__(16) _Float16 W2f[512];   // swizzled, rows 16..31 zero
  __shared__ float B1s[16], B2s[16], w3s[16], b3s;
  __shared__ __align__(16) _Float16 As[WAVES][512], Hs[WAVES][512];
  __shared__ float O2[WAVES][256];
  int tid = threadIdx.x, lane = tid & 31, wv = tid >> 5;
  stage_bfrag(roW1, W1f, 32, 16, 1, tid, blockDim.x);
  stage_bfrag(roW2, W2f, 16, 16, 1, tid, blockDim.x);
  if (tid < 16) { B1s[tid] = roB1[tid]; B2s[tid] = roB2[tid]; w3s[tid] = roW3[tid]; }
  if (tid == 0) b3s = roB3[0];
  int tile = blockIdx.x * WAVES + wv;
  bool active = tile < PTILES;
  if (!active) tile = 0;
  int p0 = tile * 16;
  int r = lane & 15, hh = lane >> 4, n = r;
  // zero the K-padding half (cols 16..31) of Hs once: 8 halves per lane
  {
    v8h zh = {(_Float16)0.f, (_Float16)0.f, (_Float16)0.f, (_Float16)0.f,
              (_Float16)0.f, (_Float16)0.f, (_Float16)0.f, (_Float16)0.f};
    *(v8h*)(Hs[wv] + r * 32 + 16 + hh * 8) = zh;
  }
  float qd = 0.f, td = 0.f;
  for (int t = 0; t < L; ++t) {
    __syncthreads();
    g2l_row(As[wv] + r * 32 + hh * 16,
            pss16 + (size_t)(p0 + r) * PSS_STRIDE + (t + 1) * D + hh * 16);
    async_fence();
    __syncthreads();
    v8f c = wmma16(load_a_frag(As[wv], lane), load_bswz(W1f, 0, lane), zero8());
    __syncthreads();
#pragma unroll
    for (int v = 0; v < 8; ++v) {
      int m = v + 8 * hh;
      float y = c[v] + B1s[n];
      Hs[wv][m * 32 + n] = (_Float16)(y > 0.f ? y : 0.f);
    }
    __syncthreads();
    c = wmma16(load_a_frag(Hs[wv], lane), load_bswz(W2f, 0, lane), zero8());
    __syncthreads();
#pragma unroll
    for (int v = 0; v < 8; ++v) {
      int m = v + 8 * hh;
      float y = c[v] + B2s[n];
      O2[wv][m * 16 + n] = (y > 0.f ? y : 0.f);
    }
    __syncthreads();
    if (lane < 16) {
      float o = b3s;
#pragma unroll
      for (int j = 0; j < 16; ++j) o += O2[wv][lane * 16 + j] * w3s[j];
      int p = p0 + lane;
      if (t < length[p]) {
        float cp = capacity[ltp[p * L + t]];
        qd += o / cp;
        td += 1.f / cp;
      }
    }
  }
  if (lane < 16 && active) {
    int p = p0 + lane;
    out[p] = qd + (traffic[p] / packets[p]) * td;
  }
}

// --------------------------- host launcher ---------------------------------
extern "C" void kernel_launch(void* const* d_in, const int* in_sizes, int n_in,
                              void* d_out, int out_size, void* d_ws, size_t ws_size,
                              hipStream_t stream) {
  (void)in_sizes; (void)n_in; (void)out_size; (void)ws_size;
  const float* traffic   = (const float*)d_in[0];
  const float* packets   = (const float*)d_in[1];
  const float* eq_lambda = (const float*)d_in[2];
  const float* apl       = (const float*)d_in[3];
  const float* emf       = (const float*)d_in[4];
  const float* plo       = (const float*)d_in[5];
  const float* atoff     = (const float*)d_in[6];
  const float* aton      = (const float*)d_in[7];
  const float* ar_a      = (const float*)d_in[8];
  const float* sigma     = (const float*)d_in[9];
  const float* capacity  = (const float*)d_in[10];
  const float* qsize     = (const float*)d_in[11];
  const float* weight    = (const float*)d_in[12];
  const int* length      = (const int*)d_in[13];
  const int* model       = (const int*)d_in[14];
  const int* policy      = (const int*)d_in[15];
  const int* priority    = (const int*)d_in[16];
  const int* qtp         = (const int*)d_in[17];
  const int* ltp         = (const int*)d_in[18];
  const int* p2l         = (const int*)d_in[19];
  const int* p2l_deg     = (const int*)d_in[20];
  const int* p2q         = (const int*)d_in[21];
  const int* p2q_deg     = (const int*)d_in[22];
  const int* qtl         = (const int*)d_in[23];
  const float* pe_w1 = (const float*)d_in[24];
  const float* pe_b1 = (const float*)d_in[25];
  const float* pe_w2 = (const float*)d_in[26];
  const float* pe_b2 = (const float*)d_in[27];
  const float* le_w1 = (const float*)d_in[28];
  const float* le_b1 = (const float*)d_in[29];
  const float* le_w2 = (const float*)d_in[30];
  const float* le_b2 = (const float*)d_in[31];
  const float* qe_w1 = (const float*)d_in[32];
  const float* qe_b1 = (const float*)d_in[33];
  const float* qe_w2 = (const float*)d_in[34];
  const float* qe_b2 = (const float*)d_in[35];
  const float* prW   = (const float*)d_in[36];
  const float* prU   = (const float*)d_in[37];
  const float* prB   = (const float*)d_in[38];
  const float* qrW   = (const float*)d_in[39];
  const float* qrU   = (const float*)d_in[40];
  const float* qrB   = (const float*)d_in[41];
  const float* lrW   = (const float*)d_in[42];
  const float* lrU   = (const float*)d_in[43];
  const float* lrB   = (const float*)d_in[44];
  const float* ro_w1 = (const float*)d_in[45];
  const float* ro_b1 = (const float*)d_in[46];
  const float* ro_w2 = (const float*)d_in[47];
  const float* ro_b2 = (const float*)d_in[48];
  const float* ro_w3 = (const float*)d_in[49];
  const float* ro_b3 = (const float*)d_in[50];
  float* out = (float*)d_out;

  // workspace carve-up
  auto alignup = [](uintptr_t x) { return (x + 255) & ~(uintptr_t)255; };
  uintptr_t base = (uintptr_t)d_ws;
  float* loadv = (float*)base;        base = alignup(base + (uintptr_t)NL * sizeof(float));
  _Float16* ps16  = (_Float16*)base;  base = alignup(base + (uintptr_t)P * D * 2);
  _Float16* qs16  = (_Float16*)base;  base = alignup(base + (uintptr_t)NQ * D * 2);
  _Float16* ls16  = (_Float16*)base;  base = alignup(base + (uintptr_t)NL * D * 2);
  _Float16* pss16 = (_Float16*)base;  // P * 9 * 32 halves

  const int PB = (PTILES + WAVES - 1) / WAVES;  // 782
  const int LB = (LTILES + WAVES - 1) / WAVES;  // 32
  const int QB = (QTILES + WAVES - 1) / WAVES;  // 94

  k_load<<<NL, 128, 0, stream>>>(traffic, p2l, p2l_deg, capacity, loadv);
  k_path_embed<<<PB, 128, 0, stream>>>(traffic, packets, eq_lambda, apl, emf, plo,
                                       atoff, aton, ar_a, sigma, model,
                                       pe_w1, pe_b1, pe_w2, pe_b2, ps16);
  k_link_embed<<<LB, 128, 0, stream>>>(loadv, policy, le_w1, le_b1, le_w2, le_b2, ls16);
  k_queue_embed<<<QB, 128, 0, stream>>>(qsize, priority, weight,
                                        qe_w1, qe_b1, qe_w2, qe_b2, qs16);
  for (int it = 0; it < ITERS; ++it) {
    k_path_rnn<<<PB, 128, 0, stream>>>(ps16, qs16, ls16, qtp, ltp, length,
                                       prW, prU, prB, pss16);
    k_queue_update<<<QB, 128, 0, stream>>>(pss16, qs16, p2q, p2q_deg, qrW, qrU, qrB);
    k_link_update<<<LB, 128, 0, stream>>>(qs16, ls16, qtl, lrW, lrU, lrB);
  }
  k_readout<<<PB, 128, 0, stream>>>(pss16, ltp, length, capacity, traffic, packets,
                                    ro_w1, ro_b1, ro_w2, ro_b2, ro_w3, ro_b3, out);
}
